// InvariantPointAttention_all_frames_70841190580545
// MI455X (gfx1250) — matrix operand order repaired and checked by reading, not compiled, not run on hardware
//
#include <hip/hip_runtime.h>
#include <hip/hip_bf16.h>

// Invariant Point Attention for MI455X (gfx1250), wave32, WMMA f32_16x16x32_f16.
// Memory-bound on z (2 x 302MB reads); all GEMMs through v_wmma; z second pass
// staged through LDS (transposed f16) for aligned ds-based B fragments.

typedef _Float16 v16h  __attribute__((ext_vector_type(16)));
typedef _Float16 half8 __attribute__((ext_vector_type(8)));
typedef float    v8f   __attribute__((ext_vector_type(8)));

#define NTOK 768
#define CSD  384
#define CZD  128
#define CHD  16
#define NHD  12
#define PQD  4
#define PVD  8

// ---------- WMMA fragment loaders ----------
// A fragment (16x32 f16, row-major source, row ptr already offset to &A[m][k0])
__device__ __forceinline__ v16h fragA(const _Float16* __restrict__ row, int lane) {
  int koff = (lane & 16) ? 8 : 0;
  half8 lo = *reinterpret_cast<const half8*>(row + koff);
  half8 hi = *reinterpret_cast<const half8*>(row + koff + 16);
  v16h f;
#pragma unroll
  for (int i = 0; i < 8; ++i) { f[i] = lo[i]; f[i + 8] = hi[i]; }
  return f;
}
// B fragment (32x16 f16, source stored N-major: row ptr = &Bt[n][k0])
__device__ __forceinline__ v16h fragB(const _Float16* __restrict__ row, int lane) {
  int koff = (lane & 16) ? 8 : 0;
  half8 lo = *reinterpret_cast<const half8*>(row + koff);
  half8 hi = *reinterpret_cast<const half8*>(row + koff + 16);
  v16h f;
#pragma unroll
  for (int i = 0; i < 8; ++i) { f[i] = lo[i]; f[i + 8] = hi[i]; }
  return f;
}
// B fragment from LDS (same N-major layout)
__device__ __forceinline__ v16h fragB_lds(const _Float16* row, int lane) {
  int koff = (lane & 16) ? 8 : 0;
  half8 lo = *reinterpret_cast<const half8*>(row + koff);
  half8 hi = *reinterpret_cast<const half8*>(row + koff + 16);
  v16h f;
#pragma unroll
  for (int i = 0; i < 8; ++i) { f[i] = lo[i]; f[i + 8] = hi[i]; }
  return f;
}
// A fragment converted on the fly from f32 source
__device__ __forceinline__ v16h fragA_f32(const float* __restrict__ row, int lane) {
  int koff = (lane & 16) ? 8 : 0;
  const float4* p0 = reinterpret_cast<const float4*>(row + koff);
  const float4* p1 = reinterpret_cast<const float4*>(row + koff + 16);
  float4 a0 = p0[0], a1 = p0[1], b0 = p1[0], b1 = p1[1];
  v16h f;
  f[0]=(_Float16)a0.x; f[1]=(_Float16)a0.y; f[2]=(_Float16)a0.z; f[3]=(_Float16)a0.w;
  f[4]=(_Float16)a1.x; f[5]=(_Float16)a1.y; f[6]=(_Float16)a1.z; f[7]=(_Float16)a1.w;
  f[8]=(_Float16)b0.x; f[9]=(_Float16)b0.y; f[10]=(_Float16)b0.z; f[11]=(_Float16)b0.w;
  f[12]=(_Float16)b1.x; f[13]=(_Float16)b1.y; f[14]=(_Float16)b1.z; f[15]=(_Float16)b1.w;
  return f;
}
__device__ __forceinline__ v8f zero8() {
  v8f z;
#pragma unroll
  for (int i = 0; i < 8; ++i) z[i] = 0.f;
  return z;
}

// ---------- utility kernels ----------
__global__ void cvt_f32_f16_k(const float* __restrict__ src, _Float16* __restrict__ dst, int n) {
  int i = blockIdx.x * blockDim.x + threadIdx.x;
  if (i < n) dst[i] = (_Float16)src[i];
}
__global__ void fill_f16_k(_Float16* __restrict__ dst, int n) {
  int i = blockIdx.x * blockDim.x + threadIdx.x;
  if (i < n) dst[i] = (_Float16)0.f;
}

// ---------- generic WMMA GEMM: C[M,N] = A[M,K](f16) * B[N,K](f16)^T + bias ----------
// Requirements (guaranteed by launch sites): M%16==0, K%32==0, (Nout/16)%4==0.
// Each wave computes a 16x64 strip: 1 A fragment feeds 4 WMMAs per K step.
__global__ void __launch_bounds__(128)
gemm_f16(const _Float16* __restrict__ A, const _Float16* __restrict__ B,
         const float* __restrict__ bias, void* __restrict__ C,
         int M, int K, int Nout,
         long long sA, long long sB, long long sC, int store_half)
{
  const int lane = threadIdx.x & 31;
  const int wave = threadIdx.x >> 5;
  const int ngroups = Nout >> 6;           // 64-wide strips
  const int Mtiles  = M >> 4;
  int wid = blockIdx.x * 4 + wave;
  if (wid >= Mtiles * ngroups) return;
  int mtile = wid / ngroups;
  int ng    = wid % ngroups;
  const _Float16* Ab = A + (long long)blockIdx.y * sA;
  const _Float16* Bb = B + (long long)blockIdx.y * sB;

  v8f acc[4];
#pragma unroll
  for (int t = 0; t < 4; ++t) acc[t] = zero8();

  const _Float16* Arow = Ab + (size_t)(mtile * 16 + (lane & 15)) * K;
  const _Float16* Brow0 = Bb + (size_t)(ng * 64 + (lane & 15)) * K;
  int Ksteps = K >> 5;
  for (int ks = 0; ks < Ksteps; ++ks) {
    v16h af = fragA(Arow + ks * 32, lane);
#pragma unroll
    for (int t = 0; t < 4; ++t) {
      v16h bf = fragB(Brow0 + (size_t)(t * 16) * K + ks * 32, lane);
      acc[t] = __builtin_amdgcn_wmma_f32_16x16x32_f16(false, af, false, bf,
                                                      (short)0, acc[t], false, false);
    }
  }
  int mrow0 = mtile * 16 + ((lane & 16) ? 8 : 0);
  int ncol  = lane & 15;
#pragma unroll
  for (int t = 0; t < 4; ++t) {
    int col = ng * 64 + t * 16 + ncol;
    float badd = bias ? bias[col] : 0.f;
#pragma unroll
    for (int r = 0; r < 8; ++r) {
      size_t idx = (size_t)(mrow0 + r) * Nout + col;
      float v = acc[t][r] + badd;
      if (store_half) ((_Float16*)C)[(long long)blockIdx.y * sC + idx] = (_Float16)v;
      else            ((float*)C)   [(long long)blockIdx.y * sC + idx] = v;
    }
  }
}

// ---------- LayerNorm rows of 384, f32 in -> f16 out ----------
__global__ void ln_k(const float* __restrict__ se, const float* __restrict__ g,
                     const float* __restrict__ b, _Float16* __restrict__ out)
{
  int row = blockIdx.x, lane = threadIdx.x;
  const float* r = se + (size_t)row * CSD;
  float s = 0.f;
  for (int c = lane; c < CSD; c += 32) s += r[c];
  for (int off = 16; off; off >>= 1) s += __shfl_xor(s, off, 32);
  float mu = s * (1.f / CSD);
  float v = 0.f;
  for (int c = lane; c < CSD; c += 32) { float d = r[c] - mu; v += d * d; }
  for (int off = 16; off; off >>= 1) v += __shfl_xor(v, off, 32);
  float rs = rsqrtf(v * (1.f / CSD) + 1e-5f);
  _Float16* o = out + (size_t)row * CSD;
  for (int c = lane; c < CSD; c += 32)
    o[c] = (_Float16)((r[c] - mu) * rs * g[c] + b[c]);
}

// ---------- mix (+frame rotations) -> Qa/Ka/vT/norm terms ----------
__global__ void mix_k(const _Float16* __restrict__ proj, const float* __restrict__ r_rot,
                      const float* __restrict__ r_trans, const float* __restrict__ mask,
                      const float* __restrict__ head_w,
                      _Float16* __restrict__ Qa, _Float16* __restrict__ Ka,
                      float* __restrict__ qn2, float* __restrict__ kn2,
                      _Float16* __restrict__ vT)
{
  int g = blockIdx.x * blockDim.x + threadIdx.x;
  if (g >= NTOK * NHD) return;
  int n = g / NHD, h = g % NHD;

  float bq_[CHD], bk_[CHD], bv_[CHD], bqp[PQD*3], bkp[PQD*3], bvp[PVD*3];
  float fq[CHD], fk[CHD], fv[CHD], fqp[PQD*3], fkp[PQD*3], fvp[PVD*3];
#pragma unroll
  for (int c = 0; c < CHD; ++c) { fq[c] = fk[c] = fv[c] = 0.f; }
#pragma unroll
  for (int d = 0; d < PQD*3; ++d) { fqp[d] = fkp[d] = 0.f; }
#pragma unroll
  for (int d = 0; d < PVD*3; ++d) { fvp[d] = 0.f; }

  for (int f = 0; f < 5; ++f) {
    const _Float16* pr = proj + (size_t)(n * 5 + f) * 1152;
    const float* R = r_rot + (size_t)(n * 5 + f) * 9;
    const float* T = r_trans + (size_t)(n * 5 + f) * 3;
    float m = (f == 0) ? 0.f : mask[n * 5 + f];
#pragma unroll
    for (int c = 0; c < CHD; ++c) {
      float tq = (float)pr[h * CHD + c];
      float tk = (float)pr[192 + h * 32 + c];
      float tv = (float)pr[192 + h * 32 + CHD + c];
      if (f == 0) { bq_[c] = tq; bk_[c] = tk; bv_[c] = tv; }
      else        { fq[c] += m * tq; fk[c] += m * tk; fv[c] += m * tv; }
    }
#pragma unroll
    for (int p = 0; p < PQD; ++p) {
      float lx = (float)pr[576 +  0 + h * PQD + p];
      float ly = (float)pr[576 + 48 + h * PQD + p];
      float lz = (float)pr[576 + 96 + h * PQD + p];
#pragma unroll
      for (int i = 0; i < 3; ++i) {
        float o = R[i*3+0]*lx + R[i*3+1]*ly + R[i*3+2]*lz + T[i];
        if (f == 0) bqp[p*3+i] = o; else fqp[p*3+i] += m * o;
      }
    }
#pragma unroll
    for (int p = 0; p < PQD + PVD; ++p) {
      float lx = (float)pr[720 +   0 + h * 12 + p];
      float ly = (float)pr[720 + 144 + h * 12 + p];
      float lz = (float)pr[720 + 288 + h * 12 + p];
#pragma unroll
      for (int i = 0; i < 3; ++i) {
        float o = R[i*3+0]*lx + R[i*3+1]*ly + R[i*3+2]*lz + T[i];
        if (p < PQD) { if (f == 0) bkp[p*3+i] = o;        else fkp[p*3+i] += m * o; }
        else         { if (f == 0) bvp[(p-PQD)*3+i] = o;  else fvp[(p-PQD)*3+i] += m * o; }
      }
    }
  }
  // weight = softmax over a singleton axis == exactly 1.0 -> mix picks frame 0
  const float w = 1.0f;
  float denom = mask[n*5+1] + mask[n*5+2] + mask[n*5+3] + mask[n*5+4] + 1e-8f;
  float iw = (1.0f - w) / denom;

  float hw = log1pf(__expf(head_w[h])) * 0.13608276348795434f; // sqrt(1/(3*PQ*9/2))
  const float s1 = 0.14433756729740643f;                       // 1/sqrt(3*CH)

  _Float16* qa = Qa + (size_t)(h * NTOK + n) * 32;
  _Float16* ka = Ka + (size_t)(h * NTOK + n) * 32;
#pragma unroll
  for (int c = 0; c < CHD; ++c) {
    float qm = w * bq_[c] + iw * fq[c];
    float km = w * bk_[c] + iw * fk[c];
    float vm = w * bv_[c] + iw * fv[c];
    qa[c] = (_Float16)(qm * s1);
    ka[c] = (_Float16)km;
    vT[(size_t)(h * 64 + c) * NTOK + n] = (_Float16)vm;
  }
  float sq = 0.f, sk = 0.f;
#pragma unroll
  for (int d = 0; d < PQD*3; ++d) {
    float qm = w * bqp[d] + iw * fqp[d];
    float km = w * bkp[d] + iw * fkp[d];
    qa[16 + d] = (_Float16)(qm * hw);
    ka[16 + d] = (_Float16)km;
    sq += qm * qm; sk += km * km;
  }
#pragma unroll
  for (int t = 28; t < 32; ++t) { qa[t] = (_Float16)0.f; ka[t] = (_Float16)0.f; }
#pragma unroll
  for (int d = 0; d < PVD*3; ++d)
    vT[(size_t)(h * 64 + 16 + d) * NTOK + n] = (_Float16)(w * bvp[d] + iw * fvp[d]);
  // columns 40..63 are zero-filled by a fill kernel before mix_k runs
  qn2[h * NTOK + n] = 0.5f * hw * sq;
  kn2[h * NTOK + n] = 0.5f * hw * sk;
}

// ---------- bias = z @ wb^T + bb  (WMMA, K=128, 12 heads padded to 16) ----------
__global__ void __launch_bounds__(128)
bias_k(const float* __restrict__ z, const _Float16* __restrict__ wbh,
       const float* __restrict__ bb, _Float16* __restrict__ biasb)
{
  int lane = threadIdx.x & 31, wave = threadIdx.x >> 5;
  long long r0 = ((long long)blockIdx.x * 4 + wave) * 16;
  v8f acc = zero8();
  const float* Arow = z + (size_t)(r0 + (lane & 15)) * CZD;
  const _Float16* Brow = wbh + (size_t)(lane & 15) * CZD;
#pragma unroll
  for (int ks = 0; ks < 4; ++ks) {
    if (ks < 3) __builtin_prefetch(Arow + (ks + 1) * 32, 0, 1);
    v16h af = fragA_f32(Arow + ks * 32, lane);
    v16h bf = fragB(Brow + ks * 32, lane);
    acc = __builtin_amdgcn_wmma_f32_16x16x32_f16(false, af, false, bf, (short)0, acc, false, false);
  }
  int nn = lane & 15;
  if (nn < NHD) {
    float badd = bb[nn];
    int m0 = (lane & 16) ? 8 : 0;
#pragma unroll
    for (int r = 0; r < 8; ++r) {
      long long row = r0 + m0 + r;
      biasb[row * NHD + nn] = (_Float16)(acc[r] + badd);
    }
  }
}

// ---------- attention logits (WMMA K=32) + softmax -> f16 attn weights ----------
__global__ void __launch_bounds__(256)
attn_k(const _Float16* __restrict__ Qa, const _Float16* __restrict__ Ka,
       const _Float16* __restrict__ biasb, const float* __restrict__ qn2,
       const float* __restrict__ kn2, const float* __restrict__ mask,
       _Float16* __restrict__ aOut)
{
  __shared__ float lg[16][NTOK];
  __shared__ float red[16][16];
  int h = blockIdx.x, i0 = blockIdx.y * 16;
  int lane = threadIdx.x & 31, wave = threadIdx.x >> 5;
  const float c3 = 0.5773502691896258f; // sqrt(1/3)

  const _Float16* Arow = Qa + (size_t)(h * NTOK + i0 + (lane & 15)) * 32;
  v16h af = fragA(Arow, lane);
  int m0 = (lane & 16) ? 8 : 0, nn = lane & 15;
  for (int jt = wave; jt < NTOK / 16; jt += 8) {
    const _Float16* Brow = Ka + (size_t)(h * NTOK + jt * 16 + (lane & 15)) * 32;
    v16h bf = fragB(Brow, lane);
    v8f acc = zero8();
    acc = __builtin_amdgcn_wmma_f32_16x16x32_f16(false, af, false, bf, (short)0, acc, false, false);
    int j = jt * 16 + nn;
    float ulj = mask[j * 5];
    float knj = kn2[h * NTOK + j];
#pragma unroll
    for (int r = 0; r < 8; ++r) {
      int m = m0 + r, i = i0 + m;
      float uli = mask[i * 5];
      float val = acc[r] + c3 * (float)biasb[((size_t)i * NTOK + j) * NHD + h]
                  - qn2[h * NTOK + i] - knj + 100000.0f * (uli * ulj - 1.0f);
      lg[m][j] = val;
    }
  }
  __syncthreads();
  int r = threadIdx.x >> 4, c0 = threadIdx.x & 15;
  float mx = -3.0e38f;
  for (int c = c0; c < NTOK; c += 16) mx = fmaxf(mx, lg[r][c]);
  red[r][c0] = mx; __syncthreads();
  for (int off = 8; off; off >>= 1) {
    if (c0 < off) red[r][c0] = fmaxf(red[r][c0], red[r][c0 + off]);
    __syncthreads();
  }
  mx = red[r][0]; __syncthreads();
  float sum = 0.f;
  for (int c = c0; c < NTOK; c += 16) {
    float e = __expf(lg[r][c] - mx);
    lg[r][c] = e; sum += e;
  }
  red[r][c0] = sum; __syncthreads();
  for (int off = 8; off; off >>= 1) {
    if (c0 < off) red[r][c0] += red[r][c0 + off];
    __syncthreads();
  }
  float inv = 1.0f / red[r][0];
  _Float16* arow = aOut + (size_t)(h * NTOK + i0 + r) * NTOK;
  for (int c = c0; c < NTOK; c += 16) arow[c] = (_Float16)(lg[r][c] * inv);
}

// ---------- o_pair[i,h,c] = sum_j a[h,i,j] * z[i,j,c] ----------
// One block per query row i: z k-tile (32x128 f32) is loaded coalesced,
// converted to f16 and stored TRANSPOSED into LDS; B fragments then come from
// two aligned 16-byte ds reads. 4 waves split the 8 column tiles (2 each).
__global__ void __launch_bounds__(128)
opair_k(const _Float16* __restrict__ aP, const float* __restrict__ z,
        float* __restrict__ opair)
{
  __shared__ _Float16 zt[CZD][32];            // [c][k], 8 KB
  int i = blockIdx.x;
  int tid = threadIdx.x;
  int lane = tid & 31, wave = tid >> 5;
  int nn = lane & 15, m = lane & 15;
  const _Float16* Abase = aP + (size_t)m * NTOK * NTOK + (size_t)i * NTOK;
  const float* zi = z + (size_t)i * NTOK * CZD;

  v8f acc[2];
  acc[0] = zero8(); acc[1] = zero8();

  for (int kt = 0; kt < NTOK / 32; ++kt) {
    __syncthreads();                          // previous tile fully consumed
    // stage 32 rows x 128 cols: 1024 float4, 8 per thread, coalesced
#pragma unroll
    for (int it = 0; it < 8; ++it) {
      int idx = tid + 128 * it;               // 0..1023
      int k  = idx >> 5;                      // 0..31
      int c4 = (idx & 31) << 2;               // 0,4,...,124
      const float* gp = zi + (size_t)(kt * 32 + k) * CZD + c4;
      if (kt + 1 < NTOK / 32)
        __builtin_prefetch(gp + 32 * CZD, 0, 1);
      float4 v = *reinterpret_cast<const float4*>(gp);
      zt[c4 + 0][k] = (_Float16)v.x;
      zt[c4 + 1][k] = (_Float16)v.y;
      zt[c4 + 2][k] = (_Float16)v.z;
      zt[c4 + 3][k] = (_Float16)v.w;
    }
    __syncthreads();
    v16h af = fragA(Abase + kt * 32, lane);
#pragma unroll
    for (int t = 0; t < 2; ++t) {
      int ct = wave * 2 + t;
      v16h bf = fragB_lds(&zt[ct * 16 + nn][0], lane);
      acc[t] = __builtin_amdgcn_wmma_f32_16x16x32_f16(false, af, false, bf,
                                                      (short)0, acc[t], false, false);
    }
  }
  int h0 = (lane & 16) ? 8 : 0;
#pragma unroll
  for (int t = 0; t < 2; ++t) {
    int ct = wave * 2 + t;
#pragma unroll
    for (int r = 0; r < 8; ++r) {
      int hh = h0 + r;
      if (hh < NHD)
        opair[((size_t)i * NHD + hh) * CZD + ct * 16 + nn] = acc[t][r];
    }
  }
}

// ---------- inverse frame-0 rotation, norms, concat -> f16 ----------
__global__ void post_k(const float* __restrict__ oall, const float* __restrict__ opair,
                       const float* __restrict__ r_rot, const float* __restrict__ r_trans,
                       _Float16* __restrict__ cat)
{
  int i = blockIdx.x, t = threadIdx.x;
  _Float16* crow = cat + (size_t)i * 2112;
  if (t < NHD * PVD) {
    int h = t >> 3, p = t & 7;
    const float* R = r_rot + (size_t)i * 45;   // frame 0
    const float* T = r_trans + (size_t)i * 15; // frame 0
    const float* src = oall + ((size_t)h * NTOK + i) * 64 + 16 + p * 3;
    float vx = src[0] - T[0], vy = src[1] - T[1], vz = src[2] - T[2];
    float ox = R[0]*vx + R[3]*vy + R[6]*vz; // R^T v
    float oy = R[1]*vx + R[4]*vy + R[7]*vz;
    float oz = R[2]*vx + R[5]*vy + R[8]*vz;
    float nrm = sqrtf(ox*ox + oy*oy + oz*oz + 1e-8f);
    crow[192 + t] = (_Float16)ox;
    crow[288 + t] = (_Float16)oy;
    crow[384 + t] = (_Float16)oz;
    crow[480 + t] = (_Float16)nrm;
  }
  for (int idx = t; idx < NHD * CHD; idx += blockDim.x) {
    int h = idx >> 4, c = idx & 15;
    crow[idx] = (_Float16)oall[((size_t)h * NTOK + i) * 64 + c];
  }
  for (int idx = t; idx < NHD * CZD; idx += blockDim.x)
    crow[576 + idx] = (_Float16)opair[(size_t)i * (NHD * CZD) + idx];
}

// ---------- host ----------
extern "C" void kernel_launch(void* const* d_in, const int* in_sizes, int n_in,
                              void* d_out, int out_size, void* d_ws, size_t ws_size,
                              hipStream_t stream)
{
  (void)in_sizes; (void)n_in; (void)out_size; (void)ws_size;
  const float* s      = (const float*)d_in[0];
  const float* z      = (const float*)d_in[1];
  const float* r_rot  = (const float*)d_in[2];
  const float* r_tr   = (const float*)d_in[3];
  const float* mask   = (const float*)d_in[4];
  const float* wq     = (const float*)d_in[5];
  const float* bq     = (const float*)d_in[6];
  const float* wkv    = (const float*)d_in[7];
  const float* bkv    = (const float*)d_in[8];
  const float* wqp    = (const float*)d_in[9];
  const float* bqp    = (const float*)d_in[10];
  const float* wkvp   = (const float*)d_in[11];
  const float* bkvp   = (const float*)d_in[12];
  const float* wb     = (const float*)d_in[13];
  const float* bb     = (const float*)d_in[14];
  const float* head_w = (const float*)d_in[15];
  const float* wout   = (const float*)d_in[16];
  const float* bout   = (const float*)d_in[17];
  const float* wexp   = (const float*)d_in[18];
  const float* bexp   = (const float*)d_in[19];
  const float* ln_g   = (const float*)d_in[20];
  const float* ln_b   = (const float*)d_in[21];
  // d_in[22]=ww, d_in[23]=bw: softmax over a length-1 axis is identically 1.0

  char* wsb = (char*)d_ws;
  size_t off = 0;
  auto alloc = [&](size_t bytes) -> void* {
    void* p = wsb + off;
    off = (off + bytes + 255) & ~(size_t)255;
    return p;
  };
  _Float16* sh     = (_Float16*)alloc(294912ull * 2);       // s f16
  _Float16* wexph  = (_Float16*)alloc(737280ull * 2);       // wexp f16
  float*    sef32  = (float*)   alloc(1474560ull * 4);      // se pre-LN
  _Float16* seh    = (_Float16*)alloc(1474560ull * 2);      // se post-LN f16
  _Float16* wcath  = (_Float16*)alloc(442368ull * 2);       // [wq;wkv;wqp;wkvp]
  float*    bcat   = (float*)   alloc(1152ull * 4);
  _Float16* projh  = (_Float16*)alloc(4423680ull * 2);      // (N*5,1152)
  _Float16* wbh    = (_Float16*)alloc(2048ull * 2);         // wb padded 16x128
  _Float16* wouth  = (_Float16*)alloc(811008ull * 2);
  _Float16* Qa     = (_Float16*)alloc(294912ull * 2);       // (H,N,32)
  _Float16* Ka     = (_Float16*)alloc(294912ull * 2);
  float*    qn2    = (float*)   alloc(9216ull * 4);
  float*    kn2    = (float*)   alloc(9216ull * 4);
  _Float16* vT     = (_Float16*)alloc(589824ull * 2);       // (H,64,N), cols 40..63 zero
  _Float16* biasb  = (_Float16*)alloc(7077888ull * 2);      // (N*N,12)
  _Float16* aPad   = (_Float16*)alloc(9437184ull * 2);      // (16,N,N)
  float*    oall   = (float*)   alloc(589824ull * 4);       // (H,N,64)
  float*    opair  = (float*)   alloc(1179648ull * 4);      // (N,H*CZ)
  _Float16* cath   = (_Float16*)alloc(1622016ull * 2);      // (N,2112)

  auto cvt = [&](const float* src, _Float16* dst, int n) {
    cvt_f32_f16_k<<<(n + 255) / 256, 256, 0, stream>>>(src, dst, n);
  };
  auto gemm = [&](const _Float16* A, const _Float16* B, const float* bias, void* C,
                  int M, int K, int Nout, int batch,
                  long long sA, long long sB, long long sC, int storeH) {
    int Mtiles = M >> 4, ngroups = Nout >> 6;
    int waves = Mtiles * ngroups, blocks = (waves + 3) >> 2;
    gemm_f16<<<dim3(blocks, batch), 128, 0, stream>>>(A, B, bias, C, M, K, Nout,
                                                      sA, sB, sC, storeH);
  };

  // 1) weight / activation converts
  cvt(s, sh, 294912);
  cvt(wexp, wexph, 737280);
  cvt(wq,   wcath,           73728);
  cvt(wkv,  wcath +  73728, 147456);
  cvt(wqp,  wcath + 221184,  55296);
  cvt(wkvp, wcath + 276480, 165888);
  cvt(wout, wouth, 811008);
  fill_f16_k<<<8, 256, 0, stream>>>(wbh, 2048);
  cvt(wb, wbh, 1536);
  fill_f16_k<<<(589824 + 255) / 256, 256, 0, stream>>>(vT, 589824); // zero incl. pad cols
  hipMemcpyAsync(bcat,       bq,   192 * 4, hipMemcpyDeviceToDevice, stream);
  hipMemcpyAsync(bcat + 192, bkv,  384 * 4, hipMemcpyDeviceToDevice, stream);
  hipMemcpyAsync(bcat + 576, bqp,  144 * 4, hipMemcpyDeviceToDevice, stream);
  hipMemcpyAsync(bcat + 720, bkvp, 432 * 4, hipMemcpyDeviceToDevice, stream);

  // 2) se = s @ wexp^T + bexp ; LayerNorm -> f16
  gemm(sh, wexph, bexp, sef32, NTOK, CSD, 1920, 1, 0, 0, 0, 0);
  ln_k<<<NTOK * 5, 32, 0, stream>>>(sef32, ln_g, ln_b, seh);

  // 3) packed projections (q|kv|qp|kvp) over all 5 frames
  gemm(seh, wcath, bcat, projh, NTOK * 5, CSD, 1152, 1, 0, 0, 0, 1);

  // 4) mix + rotations -> augmented Q/K, transposed V'
  mix_k<<<(NTOK * NHD + 127) / 128, 128, 0, stream>>>(projh, r_rot, r_tr, mask,
                                                      head_w, Qa, Ka, qn2, kn2, vT);

  // 5) pair bias  (first pass over z)
  bias_k<<<(NTOK * NTOK / 16) / 4, 128, 0, stream>>>(z, wbh, bb, biasb);

  // 6) attention logits + softmax (heads 12..15 zero-padded)
  fill_f16_k<<<(2359296 + 255) / 256, 256, 0, stream>>>(aPad + 12ull * NTOK * NTOK, 2359296);
  attn_k<<<dim3(NHD, NTOK / 16), 256, 0, stream>>>(Qa, Ka, biasb, qn2, kn2, mask, aPad);

  // 7) o and o_pt together: per-head GEMM a[h] @ V'[h]  (Nout padded to 64)
  gemm(aPad, vT, nullptr, oall, NTOK, NTOK, 64, NHD,
       (long long)NTOK * NTOK, 64ll * NTOK, 64ll * NTOK, 0);

  // 8) o_pair (second pass over z, LDS-transposed staging)
  opair_k<<<NTOK, 128, 0, stream>>>(aPad, z, opair);

  // 9) inverse rotation, norms, concat
  post_k<<<NTOK, 256, 0, stream>>>(oall, opair, r_rot, r_tr, cath);

  // 10) final: out = cat @ wout^T + bout
  gemm(cath, wouth, bout, d_out, NTOK, 2112, CSD, 1, 0, 0, 0, 0);
}
